// CognitiveCortex_26551487824567
// MI455X (gfx1250) — matrix operand order repaired and checked by reading, not compiled, not run on hardware
//
#include <hip/hip_runtime.h>
#include <hip/hip_bf16.h>
#include <stdint.h>

// ---------------------------------------------------------------------------
// MoE (router + 8 SwiGLU experts top-2 gathered + shared expert) for gfx1250.
// bf16 WMMA GEMMs, top-k gather, pre-transposed bf16 weights, double-buffered
// LDS pipeline; B-operand tiles staged by the Tensor Data Mover (TDM).
// ---------------------------------------------------------------------------

typedef __attribute__((ext_vector_type(16))) __bf16 v16bf;
typedef __attribute__((ext_vector_type(8)))  float  v8f;

union Frag16 {
    uint32_t u[8];
    v16bf    v;
};

__device__ __forceinline__ uint16_t f32_to_bf16(float f) {
    uint32_t u = __float_as_uint(f);
    uint32_t r = (u + 0x7FFFu + ((u >> 16) & 1u)) >> 16;   // RNE
    return (uint16_t)r;
}

#define NTOK 8192
#define HDIM 1024
#define IDIM 4096
#define NEXP 8

// ---------------------------------------------------------------------------
// Tensor Data Mover staging for 2-D bf16 tiles (ISA cdna5 §8.3/8.4).
// Descriptor groups built bit-exactly; all fields wave-uniform -> SGPRs.
// This toolchain's builtin takes 6 args (g0, g1, g2, g3, extra int32x8, cpol).
// ---------------------------------------------------------------------------
#if __has_builtin(__builtin_amdgcn_tensor_load_to_lds) && \
    __has_builtin(__builtin_amdgcn_s_wait_tensorcnt)
#define USE_TDM 1
typedef __attribute__((ext_vector_type(4))) unsigned int tdm4u;
typedef __attribute__((ext_vector_type(8))) int          tdm8i;
typedef __attribute__((ext_vector_type(4))) int          tdm4i;

// Load a tile of 64 rows x 32 bf16 (row stride = row_len elements) from
// gsrc (tile start) into contiguous LDS at lds_byte_addr.
__device__ __forceinline__ void tdm_load_tile_64x32(uint32_t lds_byte_addr,
                                                    const uint16_t* gsrc,
                                                    int tensor_rows, int row_len) {
    uint64_t ga = (uint64_t)(uintptr_t)gsrc;
    tdm4u g0;
    g0[0] = 1u;                                            // count=1, no gather
    g0[1] = lds_byte_addr;                                 // D#.lds_addr
    g0[2] = (uint32_t)ga;                                  // global_addr[31:0]
    g0[3] = ((uint32_t)(ga >> 32) & 0x01FFFFFFu) | 0x80000000u;  // addr[56:32] | type=2
    tdm8i g1;
    g1[0] = 0x00010000;                                    // data_size=1 (2 bytes)
    g1[1] = (int)(((uint32_t)row_len & 0xFFFFu) << 16);    // tensor_dim0 lo16
    g1[2] = (int)((((uint32_t)row_len >> 16) & 0xFFFFu) |
                  (((uint32_t)tensor_rows & 0xFFFFu) << 16));   // dim0 hi16 | dim1 lo16
    g1[3] = (int)((((uint32_t)tensor_rows >> 16) & 0xFFFFu) |
                  (32u << 16));                            // dim1 hi16 | tile_dim0=32
    g1[4] = 64;                                            // tile_dim1=64, tile_dim2=0
    g1[5] = row_len;                                       // tensor_dim0_stride[31:0]
    g1[6] = 0;                                             // stride hi, dim1_stride lo
    g1[7] = 0;
    tdm4i z4 = (tdm4i)0;                                   // groups 2/3: 2-D tile
    tdm8i z8 = (tdm8i)0;
    __builtin_amdgcn_tensor_load_to_lds(g0, g1, z4, z4, z8, 0);
}
#endif

// ---------------------------------------------------------------------------
__global__ __launch_bounds__(256) void zero_kernel(float4* __restrict__ p) {
    size_t i = (size_t)blockIdx.x * blockDim.x + threadIdx.x;
    p[i] = make_float4(0.f, 0.f, 0.f, 0.f);
}

__global__ void init_cnts_kernel(int* __restrict__ cnts) {
    if (threadIdx.x < NEXP) cnts[threadIdx.x] = 0;
}

__global__ __launch_bounds__(256) void cvt_bf16_kernel(const float* __restrict__ src,
                                                       uint16_t* __restrict__ dst) {
    size_t i = ((size_t)blockIdx.x * blockDim.x + threadIdx.x) * 4;
    float4 v = *(const float4*)(src + i);
    uint2 packed;
    packed.x = (uint32_t)f32_to_bf16(v.x) | ((uint32_t)f32_to_bf16(v.y) << 16);
    packed.y = (uint32_t)f32_to_bf16(v.z) | ((uint32_t)f32_to_bf16(v.w) << 16);
    *(uint2*)(dst + i) = packed;
}

// ---------------------------------------------------------------------------
// Transpose + convert: dst[c][r] = bf16(src[r][c]); 32x32 tiles through LDS.
// ---------------------------------------------------------------------------
__global__ __launch_bounds__(256) void tconv_kernel(const float* __restrict__ src,
                                                    uint16_t* __restrict__ dst,
                                                    int rows, int cols) {
    __shared__ uint16_t tile[32][33];
    const int r0 = blockIdx.x * 32;
    const int c0 = blockIdx.y * 32;
    {
        int tr  = threadIdx.x >> 3;
        int tc4 = (threadIdx.x & 7) * 4;
        float4 v = *(const float4*)(src + (size_t)(r0 + tr) * cols + c0 + tc4);
        tile[tr][tc4 + 0] = f32_to_bf16(v.x);
        tile[tr][tc4 + 1] = f32_to_bf16(v.y);
        tile[tr][tc4 + 2] = f32_to_bf16(v.z);
        tile[tr][tc4 + 3] = f32_to_bf16(v.w);
    }
    __syncthreads();
    {
        int wc  = threadIdx.x >> 3;
        int wr4 = (threadIdx.x & 7) * 4;
        uint2 p;
        p.x = (uint32_t)tile[wr4 + 0][wc] | ((uint32_t)tile[wr4 + 1][wc] << 16);
        p.y = (uint32_t)tile[wr4 + 2][wc] | ((uint32_t)tile[wr4 + 3][wc] << 16);
        *(uint2*)(dst + (size_t)(c0 + wc) * rows + r0 + wr4) = p;
    }
}

// ---------------------------------------------------------------------------
// Router: one wave per token; softmax + top-2 -> dense combine [N,8].
// ---------------------------------------------------------------------------
__global__ __launch_bounds__(256) void router_kernel(const float* __restrict__ X,
                                                     const float* __restrict__ Wr,
                                                     float* __restrict__ combine) {
    const int lane  = threadIdx.x & 31;
    const int wid   = threadIdx.x >> 5;
    const int token = blockIdx.x * 8 + wid;

    float acc[NEXP];
#pragma unroll
    for (int e = 0; e < NEXP; ++e) acc[e] = 0.f;

    const float* xrow = X + (size_t)token * HDIM;
    for (int h = lane; h < HDIM; h += 32) {
        float xv = xrow[h];
        const float* wr = Wr + (size_t)h * NEXP;
#pragma unroll
        for (int e = 0; e < NEXP; ++e) acc[e] += xv * wr[e];
    }
#pragma unroll
    for (int e = 0; e < NEXP; ++e) {
        float v = acc[e];
#pragma unroll
        for (int off = 16; off > 0; off >>= 1) v += __shfl_xor(v, off, 32);
        acc[e] = v;
    }
    if (lane == 0) {
        float mx = acc[0];
#pragma unroll
        for (int e = 1; e < NEXP; ++e) mx = fmaxf(mx, acc[e]);
        float p[NEXP], s = 0.f;
#pragma unroll
        for (int e = 0; e < NEXP; ++e) { p[e] = __expf(acc[e] - mx); s += p[e]; }
        float inv = __builtin_amdgcn_rcpf(s);
#pragma unroll
        for (int e = 0; e < NEXP; ++e) p[e] *= inv;
        int i1 = 0;
#pragma unroll
        for (int e = 1; e < NEXP; ++e) if (p[e] > p[i1]) i1 = e;
        int i2 = (i1 == 0) ? 1 : 0;
#pragma unroll
        for (int e = 0; e < NEXP; ++e) if (e != i1 && p[e] > p[i2]) i2 = e;
        float* c = combine + (size_t)token * NEXP;
#pragma unroll
        for (int e = 0; e < NEXP; ++e)
            c[e] = (e == i1) ? p[i1] : ((e == i2) ? p[i2] : 0.f);
    }
}

// ---------------------------------------------------------------------------
// Per-expert compacted token lists (top-2 sparsity). Order-invariant output.
// ---------------------------------------------------------------------------
__global__ __launch_bounds__(256) void build_lists_kernel(const float* __restrict__ combine,
                                                          int* __restrict__ cnts,
                                                          int* __restrict__ lists) {
    int t = blockIdx.x * 256 + threadIdx.x;
    const float* c = combine + (size_t)t * NEXP;
#pragma unroll
    for (int e = 0; e < NEXP; ++e) {
        if (c[e] > 0.f) {
            int p = atomicAdd(&cnts[e], 1);
            lists[(size_t)e * NTOK + p] = t;
        }
    }
}

// ---------------------------------------------------------------------------
// Fragment loaders, CDNA5 16-bit WMMA VGPR layouts. Operands in LDS as
// [row][16 k-pair dwords].
// ---------------------------------------------------------------------------
__device__ __forceinline__ void load_fragA(Frag16& f, const uint32_t* lds, int m, int half) {
#pragma unroll
    for (int v = 0; v < 8; ++v) {
        int kp = (v < 4) ? (half * 4 + v) : (8 + half * 4 + (v - 4));
        f.u[v] = lds[m * 16 + kp];
    }
}
__device__ __forceinline__ void load_fragB(Frag16& f, const uint32_t* lds, int n, int half) {
#pragma unroll
    for (int v = 0; v < 8; ++v)
        f.u[v] = lds[n * 16 + ((v < 4) ? (half * 4 + v) : (8 + half * 4 + (v - 4)))];
}

#define A_OFF  0
#define BG_OFF 2048
#define BU_OFF 3072
#define BUF_DW 4096

// ---------------------------------------------------------------------------
// Gate+Up GEMM (gathered rows): Hbuf[slot, n] = bf16(silu(g)*u)
// Block tile 128x64, K-step 32, double-buffered. A via register-staged VMEM
// (gathered rows), Bg/Bu via TDM (wave 0 issues, TENSORcnt + barrier).
// ---------------------------------------------------------------------------
__global__ __launch_bounds__(256) void gateup_kernel(const uint16_t* __restrict__ Xb,
                                                     const uint16_t* __restrict__ WgT,
                                                     const uint16_t* __restrict__ WuT,
                                                     uint16_t* __restrict__ Hbuf,
                                                     const int* __restrict__ lists,
                                                     const int* __restrict__ cnts,
                                                     int expert, int NI) {
    const int count = cnts ? cnts[expert] : NTOK;
    const int m0 = blockIdx.x * 128;
    if (m0 >= count) return;
    const int* list = lists ? (lists + (size_t)expert * NTOK) : nullptr;

    __shared__ uint32_t lds[2][BUF_DW];

    const int tid   = threadIdx.x;
    const int lane  = tid & 31;
    const int half  = lane >> 4;
    const int l16   = lane & 15;
    const int wid   = tid >> 5;
    const int waveM = wid & 3;
    const int waveN = wid >> 2;
    const int n0 = blockIdx.y * 64;

    const int ar = tid >> 1;
    const int ah = tid & 1;
    int mrow = m0 + ar; if (mrow >= count) mrow = count - 1;
    const int atok = list ? list[mrow] : mrow;
    const uint16_t* aSrc = Xb + (size_t)atok * HDIM + ah * 16;

    const uint16_t* gBase = WgT + (size_t)n0 * HDIM;   // tile row start (k added later)
    const uint16_t* uBase = WuT + (size_t)n0 * HDIM;

#if defined(USE_TDM)
    const uint32_t ldsBG0 = (uint32_t)(uintptr_t)&lds[0][BG_OFF];
    const uint32_t ldsBG1 = (uint32_t)(uintptr_t)&lds[1][BG_OFF];
    const uint32_t ldsBU0 = (uint32_t)(uintptr_t)&lds[0][BU_OFF];
    const uint32_t ldsBU1 = (uint32_t)(uintptr_t)&lds[1][BU_OFF];
#else
    const int bn = tid >> 2;
    const int bq = tid & 3;
    const uint16_t* gSrc = gBase + (size_t)bn * HDIM + bq * 8;
    const uint16_t* uSrc = uBase + (size_t)bn * HDIM + bq * 8;
    uint4 rg, ru;
#endif

    v8f cg[2][2] = {};
    v8f cu[2][2] = {};

    uint4 ra0, ra1;
    // prologue: stage tile 0
    {
        const uint4* s = (const uint4*)(aSrc);
        ra0 = s[0]; ra1 = s[1];
        uint4* dA = (uint4*)&lds[0][A_OFF + ar * 16 + ah * 8];
        dA[0] = ra0; dA[1] = ra1;
#if defined(USE_TDM)
        if (wid == 0) {
            tdm_load_tile_64x32(ldsBG0, gBase, NI, HDIM);
            tdm_load_tile_64x32(ldsBU0, uBase, NI, HDIM);
        }
#else
        rg = *(const uint4*)(gSrc);
        ru = *(const uint4*)(uSrc);
        *(uint4*)&lds[0][BG_OFF + bn * 16 + bq * 4] = rg;
        *(uint4*)&lds[0][BU_OFF + bn * 16 + bq * 4] = ru;
#endif
    }

    const int KSTEPS = HDIM / 32;
    for (int i = 0; i < KSTEPS; ++i) {
#if defined(USE_TDM)
        if (wid == 0) __builtin_amdgcn_s_wait_tensorcnt((short)0);   // B tile i landed
#endif
        __syncthreads();
        const int cur = i & 1;
        const bool more = (i + 1) < KSTEPS;
        if (more) {
            int kn = (i + 1) * 32;
            const uint4* s = (const uint4*)(aSrc + kn);
            ra0 = s[0]; ra1 = s[1];
#if defined(USE_TDM)
            if (wid == 0) {
                tdm_load_tile_64x32(cur ? ldsBG0 : ldsBG1, gBase + kn, NI, HDIM);
                tdm_load_tile_64x32(cur ? ldsBU0 : ldsBU1, uBase + kn, NI, HDIM);
            }
#else
            rg = *(const uint4*)(gSrc + kn);
            ru = *(const uint4*)(uSrc + kn);
            if (i + 2 < KSTEPS) {
                __builtin_prefetch(gSrc + kn + 32, 0, 3);
                __builtin_prefetch(uSrc + kn + 32, 0, 3);
            }
#endif
        }
        Frag16 fa[2], fbg[2], fbu[2];
#pragma unroll
        for (int mi = 0; mi < 2; ++mi)
            load_fragA(fa[mi], &lds[cur][A_OFF], waveM * 32 + mi * 16 + l16, half);
#pragma unroll
        for (int ni = 0; ni < 2; ++ni) {
            load_fragB(fbg[ni], &lds[cur][BG_OFF], waveN * 32 + ni * 16 + l16, half);
            load_fragB(fbu[ni], &lds[cur][BU_OFF], waveN * 32 + ni * 16 + l16, half);
        }
#pragma unroll
        for (int mi = 0; mi < 2; ++mi)
#pragma unroll
            for (int ni = 0; ni < 2; ++ni) {
                cg[mi][ni] = __builtin_amdgcn_wmma_f32_16x16x32_bf16(
                    false, fa[mi].v, false, fbg[ni].v, (short)0, cg[mi][ni], false, false);
                cu[mi][ni] = __builtin_amdgcn_wmma_f32_16x16x32_bf16(
                    false, fa[mi].v, false, fbu[ni].v, (short)0, cu[mi][ni], false, false);
            }
        if (more) {
            const int nxt = cur ^ 1;
            uint4* dA = (uint4*)&lds[nxt][A_OFF + ar * 16 + ah * 8];
            dA[0] = ra0; dA[1] = ra1;
#if !defined(USE_TDM)
            *(uint4*)&lds[nxt][BG_OFF + bn * 16 + bq * 4] = rg;
            *(uint4*)&lds[nxt][BU_OFF + bn * 16 + bq * 4] = ru;
#endif
        }
    }

#pragma unroll
    for (int mi = 0; mi < 2; ++mi)
#pragma unroll
        for (int ni = 0; ni < 2; ++ni) {
            int col = n0 + waveN * 32 + ni * 16 + l16;
#pragma unroll
            for (int r = 0; r < 8; ++r) {
                int slot = m0 + waveM * 32 + mi * 16 + half * 8 + r;
                float g = cg[mi][ni][r];
                float u = cu[mi][ni][r];
                float sig = __builtin_amdgcn_rcpf(1.f + __expf(-g));
                Hbuf[(size_t)slot * NI + col] = f32_to_bf16(g * sig * u);
            }
        }
}

// ---------------------------------------------------------------------------
// Down GEMM (gathered rows): out[tok, n] += (Hbuf @ Wd)[slot, n] * w(tok)
// ---------------------------------------------------------------------------
__global__ __launch_bounds__(256) void down_kernel(const uint16_t* __restrict__ Hbuf,
                                                   const uint16_t* __restrict__ WdT,
                                                   float* __restrict__ out,
                                                   const float* __restrict__ combine,
                                                   const int* __restrict__ lists,
                                                   const int* __restrict__ cnts,
                                                   int expert, float scale, int NI) {
    const int count = cnts ? cnts[expert] : NTOK;
    const int m0 = blockIdx.x * 128;
    if (m0 >= count) return;
    const int* list = lists ? (lists + (size_t)expert * NTOK) : nullptr;

    __shared__ uint32_t lds[2][2048 + 1024];

    const int tid   = threadIdx.x;
    const int lane  = tid & 31;
    const int half  = lane >> 4;
    const int l16   = lane & 15;
    const int wid   = tid >> 5;
    const int waveM = wid & 3;
    const int waveN = wid >> 2;
    const int n0 = blockIdx.y * 64;

    const int ar = tid >> 1;
    const int ah = tid & 1;
    const uint16_t* aSrc = Hbuf + (size_t)(m0 + ar) * NI + ah * 16;

    const uint16_t* bBase = WdT + (size_t)n0 * NI;

#if defined(USE_TDM)
    const uint32_t ldsB0 = (uint32_t)(uintptr_t)&lds[0][2048];
    const uint32_t ldsB1 = (uint32_t)(uintptr_t)&lds[1][2048];
#else
    const int bn = tid >> 2;
    const int bq = tid & 3;
    const uint16_t* bSrc = bBase + (size_t)bn * NI + bq * 8;
    uint4 rb;
#endif

    v8f c[2][2] = {};

    uint4 ra0, ra1;
    {
        const uint4* s = (const uint4*)(aSrc);
        ra0 = s[0]; ra1 = s[1];
        uint4* dA = (uint4*)&lds[0][ar * 16 + ah * 8];
        dA[0] = ra0; dA[1] = ra1;
#if defined(USE_TDM)
        if (wid == 0) tdm_load_tile_64x32(ldsB0, bBase, HDIM, NI);
#else
        rb = *(const uint4*)(bSrc);
        *(uint4*)&lds[0][2048 + bn * 16 + bq * 4] = rb;
#endif
    }

    const int KSTEPS = NI / 32;
    for (int i = 0; i < KSTEPS; ++i) {
#if defined(USE_TDM)
        if (wid == 0) __builtin_amdgcn_s_wait_tensorcnt((short)0);
#endif
        __syncthreads();
        const int cur = i & 1;
        const bool more = (i + 1) < KSTEPS;
        if (more) {
            int kn = (i + 1) * 32;
            const uint4* s = (const uint4*)(aSrc + kn);
            ra0 = s[0]; ra1 = s[1];
#if defined(USE_TDM)
            if (wid == 0) tdm_load_tile_64x32(cur ? ldsB0 : ldsB1, bBase + kn, HDIM, NI);
#else
            rb = *(const uint4*)(bSrc + kn);
            if (i + 2 < KSTEPS) __builtin_prefetch(bSrc + kn + 32, 0, 3);
#endif
        }
        Frag16 fa[2], fb[2];
#pragma unroll
        for (int mi = 0; mi < 2; ++mi)
            load_fragA(fa[mi], &lds[cur][0], waveM * 32 + mi * 16 + l16, half);
#pragma unroll
        for (int ni = 0; ni < 2; ++ni)
            load_fragB(fb[ni], &lds[cur][2048], waveN * 32 + ni * 16 + l16, half);
#pragma unroll
        for (int mi = 0; mi < 2; ++mi)
#pragma unroll
            for (int ni = 0; ni < 2; ++ni)
                c[mi][ni] = __builtin_amdgcn_wmma_f32_16x16x32_bf16(
                    false, fa[mi].v, false, fb[ni].v, (short)0, c[mi][ni], false, false);
        if (more) {
            const int nxt = cur ^ 1;
            uint4* dA = (uint4*)&lds[nxt][ar * 16 + ah * 8];
            dA[0] = ra0; dA[1] = ra1;
#if !defined(USE_TDM)
            *(uint4*)&lds[nxt][2048 + bn * 16 + bq * 4] = rb;
#endif
        }
    }

#pragma unroll
    for (int mi = 0; mi < 2; ++mi)
#pragma unroll
        for (int ni = 0; ni < 2; ++ni) {
            int col = n0 + waveN * 32 + ni * 16 + l16;
#pragma unroll
            for (int r = 0; r < 8; ++r) {
                int slot = m0 + waveM * 32 + mi * 16 + half * 8 + r;
                if (slot < count) {
                    int tok = list ? list[slot] : slot;
                    float w = scale;
                    if (combine) w *= combine[(size_t)tok * NEXP + expert];
                    out[(size_t)tok * HDIM + col] += c[mi][ni][r] * w;
                }
            }
        }
}

// ---------------------------------------------------------------------------
// Host orchestration (graph-capture safe: kernel launches only).
// ---------------------------------------------------------------------------
extern "C" void kernel_launch(void* const* d_in, const int* in_sizes, int n_in,
                              void* d_out, int out_size, void* d_ws, size_t ws_size,
                              hipStream_t stream) {
    const float* X  = (const float*)d_in[0];
    const float* Wr = (const float*)d_in[1];
    const float* Wg = (const float*)d_in[2];
    const float* Wu = (const float*)d_in[3];
    const float* Wd = (const float*)d_in[4];
    const float* Sg = (const float*)d_in[5];
    const float* Su = (const float*)d_in[6];
    const float* Sd = (const float*)d_in[7];
    float* out = (float*)d_out;

    char* ws = (char*)d_ws;
    float*    combine = (float*)ws;                 ws += (size_t)NTOK * NEXP * 4;
    int*      cnts    = (int*)ws;                   ws += 256;
    int*      lists   = (int*)ws;                   ws += (size_t)NEXP * NTOK * 4;
    uint16_t* Xb      = (uint16_t*)ws;              ws += (size_t)NTOK * HDIM * 2;
    uint16_t* Hbuf    = (uint16_t*)ws;              ws += (size_t)NTOK * IDIM * 2;
    uint16_t* WgT     = (uint16_t*)ws;              ws += (size_t)IDIM * HDIM * 2;
    uint16_t* WuT     = (uint16_t*)ws;              ws += (size_t)IDIM * HDIM * 2;
    uint16_t* WdT     = (uint16_t*)ws;              ws += (size_t)IDIM * HDIM * 2;

    dim3 blk(256);

    zero_kernel<<<(NTOK * (size_t)HDIM) / 4 / 256, blk, 0, stream>>>((float4*)out);
    init_cnts_kernel<<<1, 32, 0, stream>>>(cnts);
    router_kernel<<<NTOK / 8, blk, 0, stream>>>(X, Wr, combine);
    build_lists_kernel<<<NTOK / 256, blk, 0, stream>>>(combine, cnts, lists);
    cvt_bf16_kernel<<<(NTOK * (size_t)HDIM) / 4 / 256, blk, 0, stream>>>(X, Xb);

    dim3 grid_gu(NTOK / 128, IDIM / 64);
    dim3 grid_dn(NTOK / 128, HDIM / 64);
    dim3 grid_tw(HDIM / 32, IDIM / 32);
    dim3 grid_td(IDIM / 32, HDIM / 32);

    for (int e = 0; e < NEXP; ++e) {
        tconv_kernel<<<grid_tw, blk, 0, stream>>>(Wg + (size_t)e * HDIM * IDIM, WgT, HDIM, IDIM);
        tconv_kernel<<<grid_tw, blk, 0, stream>>>(Wu + (size_t)e * HDIM * IDIM, WuT, HDIM, IDIM);
        tconv_kernel<<<grid_td, blk, 0, stream>>>(Wd + (size_t)e * IDIM * HDIM, WdT, IDIM, HDIM);
        gateup_kernel<<<grid_gu, blk, 0, stream>>>(Xb, WgT, WuT, Hbuf, lists, cnts, e, IDIM);
        down_kernel<<<grid_dn, blk, 0, stream>>>(Hbuf, WdT, out, combine, lists, cnts, e, 1.0f, IDIM);
    }

    const int NI2 = IDIM / 2;
    dim3 grid_gs(NTOK / 128, NI2 / 64);
    dim3 grid_ts(HDIM / 32, NI2 / 32);
    dim3 grid_tds(NI2 / 32, HDIM / 32);
    tconv_kernel<<<grid_ts, blk, 0, stream>>>(Sg, WgT, HDIM, NI2);
    tconv_kernel<<<grid_ts, blk, 0, stream>>>(Su, WuT, HDIM, NI2);
    tconv_kernel<<<grid_tds, blk, 0, stream>>>(Sd, WdT, NI2, HDIM);
    gateup_kernel<<<grid_gs, blk, 0, stream>>>(Xb, WgT, WuT, Hbuf, nullptr, nullptr, 0, NI2);
    down_kernel<<<grid_dn, blk, 0, stream>>>(Hbuf, WdT, out, nullptr, nullptr, nullptr, 0, 0.5f, NI2);
}
